// AttentionPairBias_21766894256698
// MI455X (gfx1250) — matrix operand order repaired and checked by reading, not compiled, not run on hardware
//
#include <hip/hip_runtime.h>
#include <hip/hip_bf16.h>

// ---------------- types ----------------
typedef __bf16 bf16;
typedef __attribute__((ext_vector_type(16))) __bf16 v16bf;
typedef __attribute__((ext_vector_type(8)))  __bf16 v8bf;
typedef __attribute__((ext_vector_type(2)))  __bf16 v2bf;
typedef __attribute__((ext_vector_type(8)))  float   v8f;
typedef __attribute__((ext_vector_type(4)))  float   v4f;
typedef __attribute__((ext_vector_type(2)))  float   v2f;
typedef __attribute__((ext_vector_type(4)))  int     v4i;

union ABfrag { v16bf v; v8bf h[2]; unsigned u[8]; };

// ---------------- async-LDS feature detection ----------------
#if defined(__has_builtin)
#if __has_builtin(__builtin_amdgcn_global_load_async_to_lds_b128) && \
    __has_builtin(__builtin_amdgcn_s_wait_asynccnt)
#define HAVE_ASYNC_LDS 1
#endif
#endif
#ifndef HAVE_ASYNC_LDS
#define HAVE_ASYNC_LDS 0
#endif

#if HAVE_ASYNC_LDS
// one lane moves 16 bytes global -> LDS asynchronously (tracked by ASYNCcnt)
__device__ __forceinline__ void async_copy_b128(const bf16* gp, bf16* lp) {
    __builtin_amdgcn_global_load_async_to_lds_b128(
        (__attribute__((address_space(1))) v4i*)gp,
        (__attribute__((address_space(3))) v4i*)lp, 0, 0);
}
#endif

// ---------------- helpers ----------------
// scalar f32 -> bf16 bits via native fptrunc (gfx1250 has native bf16)
__device__ __forceinline__ unsigned short bf_bits(float x) {
    union { bf16 b; unsigned short s; } u; u.b = (bf16)x;
    return u.s;
}
// pack two floats into one dword of 2x bf16 (hardware packed convert)
__device__ __forceinline__ unsigned pk2(float lo, float hi) {
#if defined(__has_builtin) && __has_builtin(__builtin_amdgcn_cvt_pk_bf16_f32)
    union { v2bf v; unsigned u; } c;
    c.v = __builtin_amdgcn_cvt_pk_bf16_f32(lo, hi);
    return c.u;
#else
    union { v2bf v; unsigned u; } c;
    c.v = __builtin_convertvector((v2f){lo, hi}, v2bf);
    return c.u;
#endif
}
__device__ __forceinline__ float half_max16(float v) {   // reduce within 16-lane group
    v = fmaxf(v, __shfl_xor(v, 1, 32)); v = fmaxf(v, __shfl_xor(v, 2, 32));
    v = fmaxf(v, __shfl_xor(v, 4, 32)); v = fmaxf(v, __shfl_xor(v, 8, 32));
    return v;
}
__device__ __forceinline__ float half_sum16(float v) {
    v += __shfl_xor(v, 1, 32); v += __shfl_xor(v, 2, 32);
    v += __shfl_xor(v, 4, 32); v += __shfl_xor(v, 8, 32);
    return v;
}

// ---------------- f32 -> bf16 pre-convert (b128 in / b128 out) ----------------
__global__ __launch_bounds__(256)
void cvt_bf16_kernel(const float* __restrict__ in, unsigned int* __restrict__ out, int n8)
{
    int idx = blockIdx.x * 256 + threadIdx.x;
    if (idx >= n8) return;
    const float* p = in + (size_t)idx * 8;
    v4f a = *(const v4f*)p;
    v4f b = *(const v4f*)(p + 4);
    uint4 o;
    o.x = pk2(a[0], a[1]);  o.y = pk2(a[2], a[3]);
    o.z = pk2(b[0], b[1]);  o.w = pk2(b[2], b[3]);
    *(uint4*)(out + (size_t)idx * 4) = o;
}

// ---------------- fused NT GEMM (bf16 in):  C = act((A @ W^T + bias) * scale) -------
// A: bf16 [M,K] row-major.  W: bf16 [N,K] row-major.  Wave -> 32x64 tile (2x4 WMMA).
// B block-tile (128 x 32) staged through LDS with async global->LDS loads.
enum { MODE_BIAS = 1, MODE_SIGMOID = 2, MODE_BF16 = 4, MODE_TRANS = 8 };

template <int MODE>
__global__ __launch_bounds__(128)
void gemm_nt_kernel(const bf16* __restrict__ A, const bf16* __restrict__ W,
                    const float* __restrict__ bias, float scale,
                    void* __restrict__ outp, int M, int N, int K)
{
#if HAVE_ASYNC_LDS
    __shared__ bf16 ldsB[128 * 32];
#endif
    const int lane = threadIdx.x & 31;
    const int wave = threadIdx.x >> 5;
    const int l16  = lane & 15;
    const int half = lane >> 4;
    const int nBlkN = N >> 7;                      // 128-wide block tiles
    const int bm = (blockIdx.x / nBlkN) << 6;      // 64 rows / block
    const int bn = (blockIdx.x % nBlkN) << 7;
    const int wm = bm + ((wave >> 1) << 5);        // 32 rows / wave
    const int wn = bn + ((wave & 1) << 6);         // 64 cols / wave

    v8f c[2][4];
#pragma unroll
    for (int i = 0; i < 2; ++i)
#pragma unroll
        for (int j = 0; j < 4; ++j) c[i][j] = (v8f){0.f,0.f,0.f,0.f,0.f,0.f,0.f,0.f};

    for (int kk = 0; kk < K; kk += 32) {
#if HAVE_ASYNC_LDS
        {   // cooperative async stage of B tile: rows bn..bn+127, k = kk..kk+31
            const int t   = (int)threadIdx.x;
            const int row = t >> 2;                // 0..31
            const int ch  = t & 3;                 // 16-byte chunk within row
#pragma unroll
            for (int i = 0; i < 4; ++i) {
                const bf16* gp = W + (size_t)(bn + row + 32*i) * K + kk + ch * 8;
                bf16* lp = &ldsB[(row + 32*i) * 32 + ch * 8];
                async_copy_b128(gp, lp);
            }
            __builtin_amdgcn_s_wait_asynccnt(0);
            __syncthreads();
        }
#endif
        ABfrag a[2];
        v16bf b[4];
#pragma unroll
        for (int sm = 0; sm < 2; ++sm) {
            const bf16* ap = A + (size_t)(wm + 16*sm + l16) * K + kk + 16*half;
            a[sm].h[0] = *(const v8bf*)ap;         // K = kk + 16*half + 0..7
            a[sm].h[1] = *(const v8bf*)(ap + 8);   // K = kk + 16*half + 8..15
        }
#pragma unroll
        for (int sn = 0; sn < 4; ++sn) {
#if HAVE_ASYNC_LDS
            b[sn] = *(const v16bf*)&ldsB[((wn - bn) + 16*sn + l16) * 32 + 16*half];
#else
            b[sn] = *(const v16bf*)(W + (size_t)(wn + 16*sn + l16) * K + kk + 16*half);
#endif
        }
#pragma unroll
        for (int sm = 0; sm < 2; ++sm)
#pragma unroll
            for (int sn = 0; sn < 4; ++sn)
                c[sm][sn] = __builtin_amdgcn_wmma_f32_16x16x32_bf16(
                    false, a[sm].v, false, b[sn], (short)0, c[sm][sn], false, false);
#if HAVE_ASYNC_LDS
        __syncthreads();   // protect LDS tile before next iteration overwrites
#endif
    }

#pragma unroll
    for (int sm = 0; sm < 2; ++sm)
#pragma unroll
        for (int sn = 0; sn < 4; ++sn) {
            const int n = wn + 16*sn + l16;
            const float bv = (MODE & MODE_BIAS) ? bias[n] : 0.f;
            float tv[8];
#pragma unroll
            for (int r = 0; r < 8; ++r) {
                float v = (c[sm][sn][r] + bv) * scale;
                if (MODE & MODE_SIGMOID) v = 1.f / (1.f + __expf(-v));
                tv[r] = v;
            }
            if (MODE & MODE_TRANS) {
                // 8 consecutive rows per lane -> packed contiguous stores
                const size_t base = (size_t)n * M + wm + 16*sm + 8*half;
                if (MODE & MODE_BF16) {
                    uint4 pk;
                    pk.x = pk2(tv[0], tv[1]); pk.y = pk2(tv[2], tv[3]);
                    pk.z = pk2(tv[4], tv[5]); pk.w = pk2(tv[6], tv[7]);
                    *(uint4*)((unsigned short*)outp + base) = pk;
                } else {
                    float* o = (float*)outp + base;
                    *(v4f*)o       = (v4f){tv[0], tv[1], tv[2], tv[3]};
                    *(v4f*)(o + 4) = (v4f){tv[4], tv[5], tv[6], tv[7]};
                }
            } else {
#pragma unroll
                for (int r = 0; r < 8; ++r) {
                    const int m = wm + 16*sm + r + 8*half;
                    if (MODE & MODE_BF16)
                        ((unsigned short*)outp)[(size_t)m * N + n] = bf_bits(tv[r]);
                    else
                        ((float*)outp)[(size_t)m * N + n] = tv[r];
                }
            }
        }
}

// ---------------- pair bias via WMMA: biasT[h,j,i] = LN(z[i,j,:]) . wz[h,:] --------
__global__ __launch_bounds__(256)
void pair_bias_kernel(const float* __restrict__ z, const float* __restrict__ zw,
                      const float* __restrict__ zb, const bf16* __restrict__ wzb,
                      float* __restrict__ biasT)
{
    const int lane = threadIdx.x & 31;
    const int wave = threadIdx.x >> 5;
    const int l16  = lane & 15;
    const int half = lane >> 4;
    const size_t p0 = ((size_t)blockIdx.x * 8 + wave) * 16;

    // load this lane's 64 z values of row (p0+l16) in A-fragment order
    float x[4][16];
    const float* zr = z + (p0 + l16) * 128;
    float sum = 0.f, sq = 0.f;
#pragma unroll
    for (int ks = 0; ks < 4; ++ks) {
        const float* c0 = zr + 32*ks + 8*half;
        const float* c1 = c0 + 16;
        v4f a0 = *(const v4f*)c0, a1 = *(const v4f*)(c0 + 4);
        v4f b0 = *(const v4f*)c1, b1 = *(const v4f*)(c1 + 4);
#pragma unroll
        for (int e = 0; e < 4; ++e) {
            x[ks][e]      = a0[e]; x[ks][4 + e]  = a1[e];
            x[ks][8 + e]  = b0[e]; x[ks][12 + e] = b1[e];
        }
#pragma unroll
        for (int e = 0; e < 16; ++e) { sum += x[ks][e]; sq += x[ks][e] * x[ks][e]; }
    }
    sum += __shfl_xor(sum, 16, 32);
    sq  += __shfl_xor(sq, 16, 32);
    const float mu = sum * (1.f / 128.f);
    const float var = sq * (1.f / 128.f) - mu * mu;
    const float rs = rsqrtf(var + 1e-5f);

    // normalize + LN affine -> bf16 A-fragments (packed dword writes)
    ABfrag a[4];
#pragma unroll
    for (int ks = 0; ks < 4; ++ks) {
        const float* wr = zw + 32*ks + 8*half;
        const float* br = zb + 32*ks + 8*half;
        v4f w0 = *(const v4f*)wr,        w1 = *(const v4f*)(wr + 4);
        v4f w2 = *(const v4f*)(wr + 16), w3 = *(const v4f*)(wr + 20);
        v4f q0 = *(const v4f*)br,        q1 = *(const v4f*)(br + 4);
        v4f q2 = *(const v4f*)(br + 16), q3 = *(const v4f*)(br + 20);
        float zn[16];
#pragma unroll
        for (int e = 0; e < 4; ++e) {
            zn[e]      = (x[ks][e]      - mu) * rs * w0[e] + q0[e];
            zn[4 + e]  = (x[ks][4 + e]  - mu) * rs * w1[e] + q1[e];
            zn[8 + e]  = (x[ks][8 + e]  - mu) * rs * w2[e] + q2[e];
            zn[12 + e] = (x[ks][12 + e] - mu) * rs * w3[e] + q3[e];
        }
#pragma unroll
        for (int j = 0; j < 8; ++j) a[ks].u[j] = pk2(zn[2*j], zn[2*j + 1]);
    }
    // project to 16 heads: [16 pairs x 128] @ wz^T[128 x 16]
    v8f c = (v8f){0.f,0.f,0.f,0.f,0.f,0.f,0.f,0.f};
#pragma unroll
    for (int ks = 0; ks < 4; ++ks) {
        ABfrag bw;
        bw.h[0] = *(const v8bf*)(wzb + (size_t)l16 * 128 + 32*ks + 8*half);
        bw.h[1] = *(const v8bf*)(wzb + (size_t)l16 * 128 + 32*ks + 16 + 8*half);
        c = __builtin_amdgcn_wmma_f32_16x16x32_bf16(false, a[ks].v, false, bw.v,
                                                    (short)0, c, false, false);
    }
    // scatter to transposed layout biasT[h][j][i]
#pragma unroll
    for (int r = 0; r < 8; ++r) {
        const size_t p = p0 + r + 8*half;
        biasT[((size_t)l16 << 20) + (p & 1023) * 1024 + (p >> 10)] = c[r];
    }
}

// ---------------- flash attention with pair bias ----------------
// q,k: bf16 [S, H*64]; vT: bf16 [H*64, S]; biasT: f32 [H,S(j),S(i)];
// gT: f32 [H*64, S] (transposed gate). one wave = (head, 16 queries); 64 keys/iter.
__global__ __launch_bounds__(256)
void attn_kernel(const bf16* __restrict__ q, const bf16* __restrict__ k,
                 const bf16* __restrict__ vT, const float* __restrict__ biasT,
                 const float* __restrict__ gT, unsigned short* __restrict__ og)
{
    __shared__ unsigned short plds[8][16 * 64];
    const int lane = threadIdx.x & 31;
    const int wave = threadIdx.x >> 5;
    const int l16  = lane & 15;
    const int half = lane >> 4;
    const int wid  = blockIdx.x * 8 + wave;
    const int h    = wid >> 6;
    const int qb   = (wid & 63) << 4;
    const int S = 1024, DS = 1024, HD = 64;

    // preload Q A-fragments (K = 0..31, 32..63)
    ABfrag qa[2];
#pragma unroll
    for (int f = 0; f < 2; ++f) {
        const bf16* qp = q + (size_t)(qb + l16) * DS + h * HD + 32*f + 16*half;
        qa[f].h[0] = *(const v8bf*)qp;
        qa[f].h[1] = *(const v8bf*)(qp + 8);
    }

    v8f acc[4];
#pragma unroll
    for (int dt = 0; dt < 4; ++dt) acc[dt] = (v8f){0.f,0.f,0.f,0.f,0.f,0.f,0.f,0.f};
    float mr[8], lr[8];
#pragma unroll
    for (int r = 0; r < 8; ++r) { mr[r] = -1e30f; lr[r] = 0.f; }

    for (int j0 = 0; j0 < S; j0 += 64) {
        // scores: 4 tiles of 16 keys (1/sqrt(hd) already folded into q)
        v8f sc[4];
#pragma unroll
        for (int nt = 0; nt < 4; ++nt) {
            sc[nt] = (v8f){0.f,0.f,0.f,0.f,0.f,0.f,0.f,0.f};
#pragma unroll
            for (int ks = 0; ks < 2; ++ks) {
                const bf16* kp = k + (size_t)(j0 + 16*nt + l16) * DS + h * HD + 32*ks + 16*half;
                v16bf kb = *(const v16bf*)kp;
                sc[nt] = __builtin_amdgcn_wmma_f32_16x16x32_bf16(
                    false, qa[ks].v, false, kb, (short)0, sc[nt], false, false);
            }
            const float* bp = biasT + ((size_t)h << 20)
                            + (size_t)(j0 + 16*nt + l16) * S + qb + 8*half;
            v4f b0 = *(const v4f*)bp;
            v4f b1 = *(const v4f*)(bp + 4);
#pragma unroll
            for (int r = 0; r < 4; ++r) { sc[nt][r] += b0[r]; sc[nt][4 + r] += b1[r]; }
        }
        // online softmax per row
#pragma unroll
        for (int r = 0; r < 8; ++r) {
            float rm = fmaxf(fmaxf(sc[0][r], sc[1][r]), fmaxf(sc[2][r], sc[3][r]));
            rm = half_max16(rm);
            float mn = fmaxf(mr[r], rm);
            float alpha = __expf(mr[r] - mn);
            float p0 = __expf(sc[0][r] - mn);
            float p1 = __expf(sc[1][r] - mn);
            float p2 = __expf(sc[2][r] - mn);
            float p3 = __expf(sc[3][r] - mn);
            float rsum = half_sum16((p0 + p1) + (p2 + p3));
            lr[r] = lr[r] * alpha + rsum;
            mr[r] = mn;
#pragma unroll
            for (int dt = 0; dt < 4; ++dt) acc[dt][r] *= alpha;
            const int row = r + 8 * half;
            plds[wave][row * 64 + l16]      = bf_bits(p0);
            plds[wave][row * 64 + 16 + l16] = bf_bits(p1);
            plds[wave][row * 64 + 32 + l16] = bf_bits(p2);
            plds[wave][row * 64 + 48 + l16] = bf_bits(p3);
        }
        // re-read P (16x64) as two A-fragments (C-layout -> A-layout via LDS)
        ABfrag pa[2];
        const unsigned short* pr = &plds[wave][l16 * 64];
#pragma unroll
        for (int f = 0; f < 2; ++f) {
            pa[f].h[0] = *(const v8bf*)(pr + 32*f + 16*half);
            pa[f].h[1] = *(const v8bf*)(pr + 32*f + 16*half + 8);
        }
        // O += P V
#pragma unroll
        for (int dt = 0; dt < 4; ++dt)
#pragma unroll
            for (int f = 0; f < 2; ++f) {
                const bf16* vp = vT + (size_t)(h * HD + 16*dt + l16) * S
                               + j0 + 32*f + 16*half;
                v16bf vb = *(const v16bf*)vp;
                acc[dt] = __builtin_amdgcn_wmma_f32_16x16x32_bf16(
                    false, pa[f].v, false, vb, (short)0, acc[dt], false, false);
            }
    }
    // normalize, gate (transposed gate -> contiguous loads), store bf16
    v4f gv[4][2];
#pragma unroll
    for (int dt = 0; dt < 4; ++dt) {
        const float* gp = gT + (size_t)(h * HD + 16*dt + l16) * S + qb + 8*half;
        gv[dt][0] = *(const v4f*)gp;
        gv[dt][1] = *(const v4f*)(gp + 4);
    }
#pragma unroll
    for (int r = 0; r < 8; ++r) {
        const float inv = 1.0f / lr[r];
        const int i = qb + r + 8 * half;
#pragma unroll
        for (int dt = 0; dt < 4; ++dt) {
            const int n = h * HD + 16*dt + l16;
            float v = acc[dt][r] * inv * gv[dt][r >> 2][r & 3];
            og[(size_t)i * DS + n] = bf_bits(v);
        }
    }
}

// ---------------- launch ----------------
extern "C" void kernel_launch(void* const* d_in, const int* in_sizes, int n_in,
                              void* d_out, int out_size, void* d_ws, size_t ws_size,
                              hipStream_t stream)
{
    (void)in_sizes; (void)n_in; (void)out_size; (void)ws_size;
    const float* s  = (const float*)d_in[0];
    const float* z  = (const float*)d_in[1];
    const float* wq = (const float*)d_in[2];
    const float* bq = (const float*)d_in[3];
    const float* wk = (const float*)d_in[4];
    const float* wv = (const float*)d_in[5];
    const float* wg = (const float*)d_in[6];
    const float* zw = (const float*)d_in[7];
    const float* zb = (const float*)d_in[8];
    const float* wz = (const float*)d_in[9];
    const float* wo = (const float*)d_in[10];
    float* out = (float*)d_out;

    char* ws = (char*)d_ws;
    const size_t MB = 1048576;
    unsigned short* s_bf  = (unsigned short*)(ws + 0  * MB);  // 2 MB each
    unsigned short* wq_bf = (unsigned short*)(ws + 2  * MB);
    unsigned short* wk_bf = (unsigned short*)(ws + 4  * MB);
    unsigned short* wv_bf = (unsigned short*)(ws + 6  * MB);
    unsigned short* wg_bf = (unsigned short*)(ws + 8  * MB);
    unsigned short* wo_bf = (unsigned short*)(ws + 10 * MB);
    unsigned short* wz_bf = (unsigned short*)(ws + 12 * MB);  // 4 KB
    unsigned short* qbuf  = (unsigned short*)(ws + 13 * MB);  // 2 MB
    unsigned short* kbuf  = (unsigned short*)(ws + 15 * MB);
    unsigned short* vTbuf = (unsigned short*)(ws + 17 * MB);
    float*          gTbuf = (float*)         (ws + 19 * MB);  // 4 MB
    unsigned short* ogbuf = (unsigned short*)(ws + 23 * MB);  // 2 MB
    float*          biasT = (float*)         (ws + 25 * MB);  // 64 MB

    const int M = 1024, N = 1024, K = 1024;
    const int n8 = (1024 * 1024) / 8;

    // pre-convert activations & weights to bf16
    cvt_bf16_kernel<<<512, 256, 0, stream>>>(s,  (unsigned*)s_bf,  n8);
    cvt_bf16_kernel<<<512, 256, 0, stream>>>(wq, (unsigned*)wq_bf, n8);
    cvt_bf16_kernel<<<512, 256, 0, stream>>>(wk, (unsigned*)wk_bf, n8);
    cvt_bf16_kernel<<<512, 256, 0, stream>>>(wv, (unsigned*)wv_bf, n8);
    cvt_bf16_kernel<<<512, 256, 0, stream>>>(wg, (unsigned*)wg_bf, n8);
    cvt_bf16_kernel<<<512, 256, 0, stream>>>(wo, (unsigned*)wo_bf, n8);
    cvt_bf16_kernel<<<1, 256, 0, stream>>>(wz, (unsigned*)wz_bf, 2048 / 8);

    dim3 gG(128), tG(128);
    // q = (s@wq.T + bq) * hd^-0.5 -> bf16 [S, H*64]
    gemm_nt_kernel<MODE_BF16 | MODE_BIAS><<<gG, tG, 0, stream>>>(
        (const bf16*)s_bf, (const bf16*)wq_bf, bq, 0.125f, qbuf, M, N, K);
    // k = s@wk.T -> bf16
    gemm_nt_kernel<MODE_BF16><<<gG, tG, 0, stream>>>(
        (const bf16*)s_bf, (const bf16*)wk_bf, nullptr, 1.0f, kbuf, M, N, K);
    // v = s@wv.T -> bf16 transposed [H*64, S]
    gemm_nt_kernel<MODE_BF16 | MODE_TRANS><<<gG, tG, 0, stream>>>(
        (const bf16*)s_bf, (const bf16*)wv_bf, nullptr, 1.0f, vTbuf, M, N, K);
    // gT = sigmoid(s@wg.T) -> f32 transposed [H*64, S]
    gemm_nt_kernel<MODE_SIGMOID | MODE_TRANS><<<gG, tG, 0, stream>>>(
        (const bf16*)s_bf, (const bf16*)wg_bf, nullptr, 1.0f, gTbuf, M, N, K);
    // biasT[h,j,i] = LN(z) . wz   (WMMA projection)
    pair_bias_kernel<<<8192, 256, 0, stream>>>(z, zw, zb, (const bf16*)wz_bf, biasT);
    // flash attention + gate -> bf16 [S, H*64]
    attn_kernel<<<128, 256, 0, stream>>>((const bf16*)qbuf, (const bf16*)kbuf,
                                         (const bf16*)vTbuf, biasT, gTbuf, ogbuf);
    // out = og @ wo.T -> f32
    gemm_nt_kernel<0><<<gG, tG, 0, stream>>>(
        (const bf16*)ogbuf, (const bf16*)wo_bf, nullptr, 1.0f, out, M, N, K);
}
